// TAG_LN_EHUB_70574902608026
// MI455X (gfx1250) — matrix operand order repaired and checked by reading, not compiled, not run on hardware
//
#include <hip/hip_runtime.h>
#include <hip/hip_bf16.h>

// ---------- vector types (trivial, union-safe) ----------
typedef __attribute__((ext_vector_type(4)))  float        f32x4;
typedef __attribute__((ext_vector_type(4)))  unsigned int u32x4;
typedef __attribute__((ext_vector_type(8)))  float        v8f;
typedef __attribute__((ext_vector_type(16))) __bf16       v16bf;

union Frag16 { v16bf v; f32x4 f4[2]; };

// =====================================================================
// Graph preprocessing: degree, norm, CSR-by-destination
// =====================================================================
__global__ __launch_bounds__(256) void deg_kernel(const int* __restrict__ col,
                                                  float* __restrict__ deg, int E) {
    int e = blockIdx.x * 256 + threadIdx.x;
    if (e < E) atomicAdd(&deg[col[e]], 1.0f);
}

__global__ __launch_bounds__(256) void dis_kernel(const float* __restrict__ deg,
                                                  float* __restrict__ dis, int n) {
    int i = blockIdx.x * 256 + threadIdx.x;
    if (i >= n) return;
    float d = deg[i];
    dis[i] = (d > 0.0f) ? rsqrtf(fmaxf(d, 1.0f)) : 0.0f;
}

// single-block exclusive scan of (int)deg -> ptr ; n up to ~100K
__global__ __launch_bounds__(1024) void scan_kernel(const float* __restrict__ deg,
                                                    int* __restrict__ ptr, int n) {
    __shared__ int part[1024];
    int t = threadIdx.x;
    int chunk = (n + 1023) >> 10;
    int s = t * chunk;
    int e = s + chunk; if (e > n) e = n;
    int sum = 0;
    for (int i = s; i < e; ++i) sum += (int)deg[i];
    part[t] = sum;
    __syncthreads();
    for (int off = 1; off < 1024; off <<= 1) {
        int v = (t >= off) ? part[t - off] : 0;
        __syncthreads();
        part[t] += v;
        __syncthreads();
    }
    int run = (t > 0) ? part[t - 1] : 0;
    for (int i = s; i < e; ++i) { ptr[i] = run; run += (int)deg[i]; }
}

__global__ __launch_bounds__(256) void scatter_kernel(const int* __restrict__ row,
                                                      const int* __restrict__ col,
                                                      const float* __restrict__ dis,
                                                      int* __restrict__ ptr_next,
                                                      int* __restrict__ csr_src,
                                                      float* __restrict__ csr_w, int E) {
    int e = blockIdx.x * 256 + threadIdx.x;
    if (e >= E) return;
    int r = row[e], c = col[e];
    int pos = atomicAdd(&ptr_next[c], 1);
    csr_src[pos] = r;
    csr_w[pos]   = dis[r] * dis[c];
}

// =====================================================================
// SpMM hops: out[v] = sum_{e: dst=v} w_e * in[src_e]   (node-parallel, no atomics)
// =====================================================================
__global__ __launch_bounds__(256) void spmm_f32(const float* __restrict__ hin,
                                                float* __restrict__ hout,
                                                const int* __restrict__ ptr,
                                                const float* __restrict__ deg,
                                                const int* __restrict__ csr_src,
                                                const float* __restrict__ csr_w,
                                                int n, int D) {
    int idx = blockIdx.x * 256 + threadIdx.x;
    if (idx >= n * D) return;
    int node = idx / D, f = idx % D;
    int s = ptr[node];
    int c = (int)deg[node];
    float acc = 0.0f;
    for (int i = 0; i < c; ++i)
        acc += csr_w[s + i] * hin[(size_t)csr_src[s + i] * D + f];
    hout[idx] = acc;
}

template <int D>
__global__ __launch_bounds__(256) void spmm_bf16(const __bf16* __restrict__ hin,
                                                 __bf16* __restrict__ hout,
                                                 const int* __restrict__ ptr,
                                                 const float* __restrict__ deg,
                                                 const int* __restrict__ csr_src,
                                                 const float* __restrict__ csr_w,
                                                 int n) {
    int idx = blockIdx.x * 256 + threadIdx.x;
    if (idx >= n * D) return;
    int node = idx / D, f = idx % D;
    int s = ptr[node];
    int c = (int)deg[node];
    float acc = 0.0f;
    for (int i = 0; i < c; ++i)
        acc += csr_w[s + i] * (float)hin[(size_t)csr_src[s + i] * D + f];
    hout[idx] = (__bf16)acc;
}

// =====================================================================
// Weight packing into WMMA 16x16x32 bf16 B-fragment layout.
// Layout: [kb][nb][lane][e], lane<16: K=kb*32+(e>>3)*16+(e&7),N=nb*16+lane ; lane>=16: K=+8
// =====================================================================
__global__ __launch_bounds__(256) void packW(const float* __restrict__ W, // [D][128]
                                             __bf16* __restrict__ dst,
                                             int D, int kbPerMat) {
    int idx = blockIdx.x * 256 + threadIdx.x;
    int total = kbPerMat << 12;                 // kbPerMat * 8 * 32 * 16
    if (idx >= total) return;
    int e  = idx & 15;
    int l  = (idx >> 4) & 31;
    int nb = (idx >> 9) & 7;
    int kb = idx >> 12;
    int k  = kb * 32 + (e >> 3) * 16 + (l >> 4) * 8 + (e & 7);
    int nc = nb * 16 + (l & 15);
    float v = (k < D) ? W[k * 128 + nc] : 0.0f;
    dst[idx] = (__bf16)v;
}

// A-fragment loader: 16-bit A 16x32 layout; two 16B chunks per lane,
// upper K-half zero-padded when the matrix K (=D) ends inside the block.
__device__ __forceinline__ Frag16 loadA(const __bf16* Am, int kbase, int D, int o) {
    Frag16 a;
    a.f4[0] = *(const f32x4*)(Am + kbase + o);              // K [o, o+8)
    if (kbase + 32 <= D)
        a.f4[1] = *(const f32x4*)(Am + kbase + 16 + o);     // K [16+o, 24+o)
    else
        a.f4[1] = (f32x4){0.f, 0.f, 0.f, 0.f};
    return a;
}

// =====================================================================
// WMMA GEMM:  out = relu?( sum_mat A_mat @ W_mat + bias )
// A: nmats slots of [n x D] bf16 row-major (contiguous slots)
// Wp: packed per-mat fragments (KB*4096 bf16 each), staged into LDS via
//     GLOBAL_LOAD_ASYNC_TO_LDS_B128 (ASYNCcnt path).
// One wave -> 16 rows x 128 cols. KB compile-time => full unroll:
// all A frags + all 8 B frags of a K-block live in registers, WMMAs burst.
// =====================================================================
template <int KB>
__global__ __launch_bounds__(256, 1) void gemm_tag(const __bf16* __restrict__ A,
                                                   const __bf16* __restrict__ Wp,
                                                   const float*  __restrict__ bias,
                                                   __bf16* __restrict__ outB,
                                                   int n, int D, int nmats, int relu) {
    extern __shared__ char smem[];
    __bf16* lw = (__bf16*)smem;
    const int matElems = KB << 12;              // KB * 4096
    const unsigned ldsBase = __builtin_amdgcn_groupstaticsize(); // dyn-LDS start

    int wave = threadIdx.x >> 5;
    int lane = threadIdx.x & 31;
    int m0   = (blockIdx.x * 8 + wave) * 16;
    int hi   = lane >> 4;                       // K-half selector
    int m    = m0 + (lane & 15);
    int mc   = (m < n) ? m : (n - 1);           // clamped load row

    v8f acc[8];
#pragma unroll
    for (int nb = 0; nb < 8; ++nb) acc[nb] = (v8f){0,0,0,0,0,0,0,0};

    const int o = hi * 8;                       // per-lane K offset within block
    for (int mat = 0; mat < nmats; ++mat) {
        __syncthreads();                        // LDS reads of prev mat done
        {   // async DMA: packed weights -> LDS (no VGPR round-trip)
            const char* gbase = (const char*)(Wp + (size_t)mat * matElems);
#pragma unroll
            for (int it = 0; it < KB * 2; ++it) {
                unsigned i = it * 256u + threadIdx.x;
                unsigned loff = ldsBase + i * 16u;
                const void* gp = gbase + (size_t)i * 16u;
                asm volatile("global_load_async_to_lds_b128 %0, %1, off"
                             :: "v"(loff), "v"(gp) : "memory");
            }
            asm volatile("s_wait_asynccnt 0x0" ::: "memory");
        }
        __syncthreads();                        // staged data visible to all waves

        const __bf16* Am = A + (size_t)mat * (size_t)n * D + (size_t)mc * D;
        Frag16 a[KB];
#pragma unroll
        for (int kb = 0; kb < KB; ++kb)
            a[kb] = loadA(Am, kb * 32, D, o);

#pragma unroll
        for (int kb = 0; kb < KB; ++kb) {
            const __bf16* wb = lw + (size_t)(kb * 8) * 512 + lane * 16;
            Frag16 bf[8];
#pragma unroll
            for (int nb = 0; nb < 8; ++nb) {    // 16 ds_load_b128 in flight
                bf[nb].f4[0] = *(const f32x4*)(wb + nb * 512);
                bf[nb].f4[1] = *(const f32x4*)(wb + nb * 512 + 8);
            }
#pragma unroll
            for (int nb = 0; nb < 8; ++nb)      // WMMA burst
                acc[nb] = __builtin_amdgcn_wmma_f32_16x16x32_bf16(
                    false, a[kb].v, false, bf[nb].v, (short)0, acc[nb], false, false);
        }
    }

    // D layout: VGPR r -> lanes0-15: (M=r, N=lane) ; lanes16-31: (M=8+r, N=lane-16)
    int row0 = m0 + hi * 8;
    int ncl  = lane & 15;
#pragma unroll
    for (int nb = 0; nb < 8; ++nb) {
        int ncol = nb * 16 + ncl;
        float bv = bias[ncol];
#pragma unroll
        for (int r = 0; r < 8; ++r) {
            int rr = row0 + r;
            if (rr < n) {
                float v = acc[nb][r] + bv;
                if (relu) v = (v > 0.0f) ? v : 0.0f;
                outB[(size_t)rr * 128 + ncol] = (__bf16)v;
            }
        }
    }
}

// =====================================================================
// featureEmbedding dense part: out16 = x@W0 + xa@W1 + b   (17 -> 16)
// =====================================================================
__global__ __launch_bounds__(256) void fe_dense(const float* __restrict__ x,
                                                const float* __restrict__ xa,
                                                const float* __restrict__ fe_ws,
                                                const float* __restrict__ fe_b,
                                                __bf16* __restrict__ out16, int n) {
    int idx = blockIdx.x * 256 + threadIdx.x;
    if (idx >= n * 16) return;
    int node = idx >> 4, o = idx & 15;
    const float* W0 = fe_ws;
    const float* W1 = fe_ws + 17 * 16;
    float acc = fe_b[o];
    const float* xr  = x  + (size_t)node * 17;
    const float* xar = xa + (size_t)node * 17;
    for (int i = 0; i < 17; ++i)
        acc += xr[i] * W0[i * 16 + o] + xar[i] * W1[i * 16 + o];
    out16[idx] = (__bf16)acc;
}

// final FC: out[n] = h[n,:] . fc_w + fc_b
__global__ __launch_bounds__(256) void fc_kernel(const __bf16* __restrict__ h,
                                                 const float* __restrict__ fcw,
                                                 const float* __restrict__ fcb,
                                                 float* __restrict__ out, int n) {
    int node = blockIdx.x * 256 + threadIdx.x;
    if (node >= n) return;
    const __bf16* hr = h + (size_t)node * 128;
    float acc = fcb[0];
    for (int j = 0; j < 128; ++j) acc += (float)hr[j] * fcw[j];
    out[node] = acc;
}

// =====================================================================
// Host-side orchestration
// =====================================================================
extern "C" void kernel_launch(void* const* d_in, const int* in_sizes, int n_in,
                              void* d_out, int out_size, void* d_ws, size_t ws_size,
                              hipStream_t stream) {
    const int N = in_sizes[0] / 17;     // 100000
    const int E = in_sizes[1] / 2;      // 1600000

    const float* x        = (const float*)d_in[0];
    const int*   erow     = (const int*)d_in[1];
    const int*   ecol     = erow + E;
    const float* fe_ws    = (const float*)d_in[2];
    const float* fe_b     = (const float*)d_in[3];
    const float* conv0_ws = (const float*)d_in[4];  // [4][16][128]
    const float* conv0_b  = (const float*)d_in[5];
    const float* conv_ws  = (const float*)d_in[6];  // [3][4][128][128]
    const float* conv_b   = (const float*)d_in[7];  // [3][128]
    const float* fc_w     = (const float*)d_in[8];
    const float* fc_b     = (const float*)d_in[9];

    // ---- carve workspace ----
    char* wsb = (char*)d_ws;
    size_t off = 0;
    auto carve = [&](size_t bytes) -> void* {
        void* p = wsb + off;
        off = (off + bytes + 255) & ~(size_t)255;
        return p;
    };
    float*  deg     = (float*)carve((size_t)N * 4);
    float*  dis     = (float*)carve((size_t)N * 4);
    int*    ptr     = (int*)  carve((size_t)N * 4);
    int*    ptrn    = (int*)  carve((size_t)N * 4);
    int*    csr_src = (int*)  carve((size_t)E * 4);
    float*  csr_w   = (float*)carve((size_t)E * 4);
    float*  xa      = (float*)carve((size_t)N * 17 * 4);
    __bf16* pw      = (__bf16*)carve((size_t)(4 * 4096 + 12 * 16384) * 2);
    __bf16* bankA   = (__bf16*)carve((size_t)4 * N * 128 * 2);
    __bf16* bankB   = (__bf16*)carve((size_t)4 * N * 128 * 2);
    (void)ws_size; (void)n_in; (void)out_size;

    const int gE = (E + 255) / 256;
    const int gN = (N + 255) / 256;

    // ---- graph preprocessing ----
    hipMemsetAsync(deg, 0, (size_t)N * 4, stream);
    deg_kernel<<<gE, 256, 0, stream>>>(ecol, deg, E);
    dis_kernel<<<gN, 256, 0, stream>>>(deg, dis, N);
    scan_kernel<<<1, 1024, 0, stream>>>(deg, ptr, N);
    hipMemcpyAsync(ptrn, ptr, (size_t)N * 4, hipMemcpyDeviceToDevice, stream);
    scatter_kernel<<<gE, 256, 0, stream>>>(erow, ecol, dis, ptrn, csr_src, csr_w, E);

    // ---- pack weights (bf16 B-fragment layout) ----
    for (int k = 0; k < 4; ++k)
        packW<<<16, 256, 0, stream>>>(conv0_ws + (size_t)k * 16 * 128,
                                      pw + (size_t)k * 4096, 16, 1);
    for (int m = 0; m < 12; ++m)
        packW<<<64, 256, 0, stream>>>(conv_ws + (size_t)m * 128 * 128,
                                      pw + 4 * 4096 + (size_t)m * 16384, 128, 4);

    // ---- featureEmbedding (17 -> 16, K=1) ----
    spmm_f32<<<(N * 17 + 255) / 256, 256, 0, stream>>>(x, xa, ptr, deg, csr_src, csr_w, N, 17);
    fe_dense<<<(N * 16 + 255) / 256, 256, 0, stream>>>(x, xa, fe_ws, fe_b, bankA, N);

    // ---- conv0: TAGConv(16 -> 128, K=3) + ReLU ----
    {
        size_t S = (size_t)N * 16;
        int g16 = (N * 16 + 255) / 256;
        spmm_bf16<16><<<g16, 256, 0, stream>>>(bankA,         bankA + S,     ptr, deg, csr_src, csr_w, N);
        spmm_bf16<16><<<g16, 256, 0, stream>>>(bankA + S,     bankA + 2 * S, ptr, deg, csr_src, csr_w, N);
        spmm_bf16<16><<<g16, 256, 0, stream>>>(bankA + 2 * S, bankA + 3 * S, ptr, deg, csr_src, csr_w, N);
        gemm_tag<1><<<(N + 127) / 128, 256, 1 * 4096 * 2, stream>>>(
            bankA, pw, conv0_b, bankB, N, 16, 4, 1);
    }

    // ---- hidden TAGConv(128 -> 128, K=3) layers ----
    __bf16* cur = bankB;
    __bf16* nxt = bankA;
    const int g128 = (N * 128 + 255) / 256;
    for (int L = 0; L < 3; ++L) {
        size_t S = (size_t)N * 128;
        spmm_bf16<128><<<g128, 256, 0, stream>>>(cur,         cur + S,     ptr, deg, csr_src, csr_w, N);
        spmm_bf16<128><<<g128, 256, 0, stream>>>(cur + S,     cur + 2 * S, ptr, deg, csr_src, csr_w, N);
        spmm_bf16<128><<<g128, 256, 0, stream>>>(cur + 2 * S, cur + 3 * S, ptr, deg, csr_src, csr_w, N);
        gemm_tag<4><<<(N + 127) / 128, 256, 4 * 4096 * 2, stream>>>(
            cur, pw + 4 * 4096 + (size_t)L * 4 * 16384, conv_b + (size_t)L * 128,
            nxt, N, 128, 4, (L < 2) ? 1 : 0);
        __bf16* t = cur; cur = nxt; nxt = t;
    }

    // ---- final FC ----
    fc_kernel<<<gN, 256, 0, stream>>>(cur, fc_w, fc_b, (float*)d_out, N);
}
